// GCN_20590073217318
// MI455X (gfx1250) — matrix operand order repaired and checked by reading, not compiled, not run on hardware
//
#include <hip/hip_runtime.h>

typedef __attribute__((ext_vector_type(16))) _Float16 v16h;
typedef __attribute__((ext_vector_type(8)))  float    v8f;
typedef __attribute__((ext_vector_type(4)))  float    v4f;

#define N_NODES 1024
#define N_EDGES 65536
#define HID     64
#define STATE   1024
#define POOLED  256      // STATE/4
#define NHEADS  1024
#define ACTD    2

// workspace layout (float offsets); AGG/CNT/Y3 contiguous so one zero pass covers them
#define OFF_AGG 0
#define OFF_CNT (OFF_AGG + N_NODES*HID)      // 65536
#define OFF_Y3  (OFF_CNT + N_NODES)          // 66560
#define ZERO_N  (OFF_Y3 + POOLED)            // 66816 floats to zero
#define OFF_W2P ZERO_N                       // 64*256
#define OFF_B2P (OFF_W2P + HID*POOLED)
#define OFF_Z   (OFF_B2P + POOLED)           // 1024*256
#define OFF_A5  (OFF_Z + N_NODES*POOLED)
#define OFF_VAL (OFF_A5 + 128)

__global__ void k_zero(float* __restrict__ p, int n) {
  int i = blockIdx.x * blockDim.x + threadIdx.x;
  if (i < n) p[i] = 0.0f;
}

// Fold AvgPool1d(4) into W2/b2:  w2p[k][c] = mean(W2[k][4c..4c+3])
__global__ void k_pool_w2(const float* __restrict__ W2, const float* __restrict__ b2,
                          float* __restrict__ w2p, float* __restrict__ b2p) {
  int tid = blockIdx.x * blockDim.x + threadIdx.x;   // 16384 threads
  int k = tid >> 8, c = tid & 255;
  const float* r = W2 + (size_t)k * STATE + c * 4;
  w2p[tid] = 0.25f * (r[0] + r[1] + r[2] + r[3]);
  if (tid < POOLED) {
    const float* bb = b2 + tid * 4;
    b2p[tid] = 0.25f * (bb[0] + bb[1] + bb[2] + bb[3]);
  }
}

// One thread per (edge, hidden-k): 64 consecutive lanes share an edge -> coalesced atomics
__global__ void k_edge(const float* __restrict__ x, const long long* __restrict__ ei,
                       const float* __restrict__ W1, const float* __restrict__ b1,
                       float* __restrict__ agg, float* __restrict__ cnt) {
  int gid = blockIdx.x * blockDim.x + threadIdx.x;
  int e = gid >> 6;
  int k = gid & 63;
  int src = (int)ei[e];
  int dst = (int)ei[N_EDGES + e];
  float xi0 = x[dst * 2 + 0], xi1 = x[dst * 2 + 1];
  float xj0 = x[src * 2 + 0], xj1 = x[src * 2 + 1];
  float e2 = xj0 - xi0, e3 = xj1 - xi1;
  float h = fmaf(xi0, W1[k],
           fmaf(xi1, W1[64 + k],
           fmaf(e2,  W1[128 + k],
           fmaf(e3,  W1[192 + k], b1[k]))));
  h = h > 0.0f ? h : 0.0f;
  atomicAdd(&agg[dst * HID + k], h);
  if (k == 0) atomicAdd(&cnt[dst], 1.0f);
}

// z[1024,256] = (agg/max(cnt,1)) @ w2p + (cnt>0)*b2p  via v_wmma_f32_16x16x32_f16
// One wave per 16x16 output tile; K=64 -> 2 chained WMMAs.
__global__ void __launch_bounds__(32) k_zgemm(const float* __restrict__ agg,
                                              const float* __restrict__ cnt,
                                              const float* __restrict__ w2p,
                                              const float* __restrict__ b2p,
                                              float* __restrict__ z) {
  const int lane = threadIdx.x;           // 0..31, full wave (EXEC all ones)
  const int tm = blockIdx.x;              // 64 row tiles
  const int tn = blockIdx.y;              // 16 col tiles
  const int hi = lane >> 4;               // half-wave select
  const int mrow = tm * 16 + (lane & 15); // A-matrix: M = lane%16 for both halves
  const int ncol = tn * 16 + (lane & 15); // B/C/D: N = lane%16

  float inv = 1.0f / fmaxf(cnt[mrow], 1.0f);

  v8f acc = {};
#pragma unroll
  for (int s = 0; s < 2; ++s) {
    const int ks = s * 32;
    v16h a, b;
#pragma unroll
    for (int h = 0; h < 16; ++h) {
      int vg = h >> 1, wh = h & 1;
      // 16-bit A 16x32 layout: lanes0-15 K in {0-7,16-23}, lanes16-31 K in {8-15,24-31}
      int ka = (vg < 4) ? ((hi ? 8 : 0) + vg * 2 + wh)
                        : ((hi ? 24 : 16) + (vg - 4) * 2 + wh);
      a[h] = (_Float16)(agg[mrow * HID + ks + ka] * inv);
      // 16-bit B 32x16 layout (matches A): lanes0-15 K in {0-7,16-23}, lanes16-31 K in {8-15,24-31}
      int kb = (vg < 4) ? ((hi ? 8 : 0) + vg * 2 + wh)
                        : ((hi ? 24 : 16) + (vg - 4) * 2 + wh);
      b[h] = (_Float16)(w2p[(ks + kb) * POOLED + ncol]);
    }
    acc = __builtin_amdgcn_wmma_f32_16x16x32_f16(false, a, false, b, (short)0, acc,
                                                 false, false);
  }
  // D layout: VGPR r holds M = r (lanes 0-15) / 8+r (lanes 16-31), N = lane%16
#pragma unroll
  for (int r = 0; r < 8; ++r) {
    int m = tm * 16 + r + (hi ? 8 : 0);
    float gate = (cnt[m] > 0.0f) ? 1.0f : 0.0f;   // isolated nodes: reference yields 0 (no bias)
    z[m * POOLED + ncol] = acc[r] + gate * b2p[ncol];
  }
}

// y3[256] = z[262144] @ W3[262144,256] : the 268 MB HBM-bound stream.
// 512 blocks x 256 lanes; each lane owns a float4 column group (B128 loads, 512B/wave/instr);
// non-temporal so the one-shot 268MB stream doesn't evict the 192MB L2's useful lines.
#define W3_ROWS_PER_BLK 512
__global__ void k_gemv_w3(const float* __restrict__ z, const float* __restrict__ W3,
                          float* __restrict__ y3) {
  __shared__ float zs[W3_ROWS_PER_BLK];
  __shared__ float red[4 * 256];
  int t = threadIdx.x;          // 256
  int team = t >> 6;            // 0..3 : row interleave group
  int c4 = (t & 63) * 4;        // column base (covers all 256 columns)
  size_t base = (size_t)blockIdx.x * W3_ROWS_PER_BLK;
  for (int r = t; r < W3_ROWS_PER_BLK; r += 256) zs[r] = z[base + r];
  __syncthreads();
  v4f acc = {0.f, 0.f, 0.f, 0.f};
#pragma unroll 4
  for (int r = team; r < W3_ROWS_PER_BLK; r += 4) {
    const v4f* p = (const v4f*)(W3 + (base + r) * POOLED + c4);
    v4f w = __builtin_nontemporal_load(p);
    float zr = zs[r];
    acc.x = fmaf(zr, w.x, acc.x);
    acc.y = fmaf(zr, w.y, acc.y);
    acc.z = fmaf(zr, w.z, acc.z);
    acc.w = fmaf(zr, w.w, acc.w);
  }
  red[team * 256 + c4 + 0] = acc.x;
  red[team * 256 + c4 + 1] = acc.y;
  red[team * 256 + c4 + 2] = acc.z;
  red[team * 256 + c4 + 3] = acc.w;
  __syncthreads();
  float s = red[t] + red[256 + t] + red[512 + t] + red[768 + t];
  atomicAdd(&y3[t], s);
}

// Dense trunk + value head, one 256-thread block (tiny)
__global__ void k_trunk(const float* __restrict__ y3, const float* __restrict__ b3,
                        const float* __restrict__ W4, const float* __restrict__ b4,
                        const float* __restrict__ W5, const float* __restrict__ b5,
                        const float* __restrict__ Wv, const float* __restrict__ bv,
                        float* __restrict__ a5_out, float* __restrict__ val_out) {
  __shared__ float a3[256], a4[256], a5[128];
  int t = threadIdx.x;
  float v = y3[t] + b3[t];
  a3[t] = v > 0.0f ? v : 0.01f * v;
  __syncthreads();
  float s = b4[t];
  for (int f = 0; f < 256; ++f) s = fmaf(a3[f], W4[f * 256 + t], s);
  a4[t] = s > 0.0f ? s : 0.01f * s;
  __syncthreads();
  if (t < 128) {
    float s5 = b5[t];
    for (int f = 0; f < 256; ++f) s5 = fmaf(a4[f], W5[f * 128 + t], s5);
    float r = s5 > 0.0f ? s5 : 0.01f * s5;
    a5[t] = r;
    a5_out[t] = r;
  }
  __syncthreads();
  if (t == 0) {
    float sv = bv[0];
    for (int f = 0; f < 128; ++f) sv = fmaf(a5[f], Wv[f], sv);
    val_out[0] = sv;
  }
}

// 1024 dueling heads: q[n,a] = value + adv[n,a] - mean_a(adv[n,:]); float4 Wa reads
__global__ void k_heads(const float* __restrict__ a5, const float* __restrict__ val,
                        const float* __restrict__ Wa, const float* __restrict__ ba,
                        float* __restrict__ q) {
  __shared__ float zs[128];
  int t = threadIdx.x;
  if (t < 128) zs[t] = a5[t];
  __syncthreads();
  int n = blockIdx.x * blockDim.x + t;
  float v = val[0];
  const v4f* w = (const v4f*)(Wa + (size_t)n * 128 * ACTD);
  float a0 = ba[n * 2 + 0], a1 = ba[n * 2 + 1];
#pragma unroll 4
  for (int f4 = 0; f4 < 64; ++f4) {
    v4f ww = w[f4];                 // {Wa[n,2f,0], Wa[n,2f,1], Wa[n,2f+1,0], Wa[n,2f+1,1]}
    float z0 = zs[f4 * 2 + 0], z1 = zs[f4 * 2 + 1];
    a0 = fmaf(z0, ww.x, a0);
    a1 = fmaf(z0, ww.y, a1);
    a0 = fmaf(z1, ww.z, a0);
    a1 = fmaf(z1, ww.w, a1);
  }
  float m = 0.5f * (a0 + a1);
  q[n * 2 + 0] = v + a0 - m;
  q[n * 2 + 1] = v + a1 - m;
}

extern "C" void kernel_launch(void* const* d_in, const int* in_sizes, int n_in,
                              void* d_out, int out_size, void* d_ws, size_t ws_size,
                              hipStream_t stream) {
  const float*     x  = (const float*)d_in[0];
  const long long* ei = (const long long*)d_in[1];
  const float* W1 = (const float*)d_in[2];
  const float* b1 = (const float*)d_in[3];
  const float* W2 = (const float*)d_in[4];
  const float* b2 = (const float*)d_in[5];
  const float* W3 = (const float*)d_in[6];
  const float* b3 = (const float*)d_in[7];
  const float* W4 = (const float*)d_in[8];
  const float* b4 = (const float*)d_in[9];
  const float* W5 = (const float*)d_in[10];
  const float* b5 = (const float*)d_in[11];
  const float* Wv = (const float*)d_in[12];
  const float* bv = (const float*)d_in[13];
  const float* Wa = (const float*)d_in[14];
  const float* ba = (const float*)d_in[15];
  float* out = (float*)d_out;
  float* ws  = (float*)d_ws;

  float* agg = ws + OFF_AGG;
  float* cnt = ws + OFF_CNT;
  float* y3  = ws + OFF_Y3;
  float* w2p = ws + OFF_W2P;
  float* b2p = ws + OFF_B2P;
  float* zb  = ws + OFF_Z;
  float* a5  = ws + OFF_A5;
  float* val = ws + OFF_VAL;

  k_zero<<<(ZERO_N + 255) / 256, 256, 0, stream>>>(ws, ZERO_N);
  k_pool_w2<<<(HID * POOLED) / 256, 256, 0, stream>>>(W2, b2, w2p, b2p);
  k_edge<<<(N_EDGES * HID) / 256, 256, 0, stream>>>(x, ei, W1, b1, agg, cnt);
  k_zgemm<<<dim3(N_NODES / 16, POOLED / 16), 32, 0, stream>>>(agg, cnt, w2p, b2p, zb);
  k_gemv_w3<<<(N_NODES * POOLED) / W3_ROWS_PER_BLK, 256, 0, stream>>>(zb, W3, y3);
  k_trunk<<<1, 256, 0, stream>>>(y3, b3, W4, b4, W5, b5, Wv, bv, a5, val);
  k_heads<<<NHEADS / 256, 256, 0, stream>>>(a5, val, Wa, ba, out);
}
